// DeepMultiheadAttention_30863634989440
// MI455X (gfx1250) — matrix operand order repaired and checked by reading, not compiled, not run on hardware
//
#include <hip/hip_runtime.h>

// ---------------------------------------------------------------------------
// CDNA5 (gfx1250) DeepMultiheadAttention forward.
// - All GEMMs: V_WMMA_F32_16X16X32_BF16 (bf16 operands, f32 accumulate).
// - Linear GEMMs: workgroup LDS staging, double-buffered with
//   GLOBAL_LOAD_ASYNC_TO_LDS_B128 (ASYNCcnt).
// - Attention: s x s matrix never materialized; top-k(softmax)==top-k(logits),
//   probabilities recovered from online (rowmax, sumexp). K-tile loads are
//   software-pipelined behind the scan; scan split across both lane halves.
// ---------------------------------------------------------------------------

typedef __attribute__((ext_vector_type(16))) __bf16 v16bf;
typedef __attribute__((ext_vector_type(8)))  float  v8f;
typedef __attribute__((ext_vector_type(4)))  int    v4i;

#define DEV __device__ __forceinline__

#if __has_builtin(__builtin_amdgcn_global_load_async_to_lds_b128)
#define ASYNC_COPY 1
typedef __attribute__((address_space(1))) v4i* as1_v4i_p;  // global int4*
typedef __attribute__((address_space(3))) v4i* as3_v4i_p;  // LDS int4*
#else
#define ASYNC_COPY 0
#endif

#if __has_builtin(__builtin_amdgcn_s_wait_asynccnt)
#define WAIT_ASYNC(n) __builtin_amdgcn_s_wait_asynccnt(n)
#else
#define WAIT_ASYNC(n) asm volatile("s_wait_asynccnt %0" :: "i"(n) : "memory")
#endif

constexpr int Bsz  = 4;
constexpr int Ssz  = 2048;
constexpr int E    = 512;
constexpr int H    = 8;
constexpr int HD   = 64;          // head dim
constexpr int TK   = 8;           // TOPK
constexpr int M    = Bsz * Ssz;   // 8192 rows
constexpr int NQKV = 3 * E;       // 1536
constexpr int DV   = TK * E;      // 4096  (vals width)
constexpr int D1   = E * (TK/2);  // 2048  (hidden width)

// ---- workspace layout (bytes, all 256B aligned) ---------------------------
constexpr size_t SZ_XB   = (size_t)M * E * 2;
constexpr size_t SZ_WQKV = (size_t)NQKV * E * 2;
constexpr size_t SZ_W1   = (size_t)D1 * DV * 2;
constexpr size_t SZ_W2   = (size_t)E * D1 * 2;
constexpr size_t SZ_QB   = (size_t)M * E * 2;
constexpr size_t SZ_KB   = (size_t)M * E * 2;
constexpr size_t SZ_VF   = (size_t)M * E * 4;
constexpr size_t SZ_VALS = (size_t)M * DV * 2;

constexpr size_t OFF_XB   = 0;
constexpr size_t OFF_WQKV = OFF_XB   + SZ_XB;
constexpr size_t OFF_W1   = OFF_WQKV + SZ_WQKV;
constexpr size_t OFF_W2   = OFF_W1   + SZ_W1;
constexpr size_t OFF_QB   = OFF_W2   + SZ_W2;
constexpr size_t OFF_KB   = OFF_QB   + SZ_QB;
constexpr size_t OFF_VF   = OFF_KB   + SZ_KB;
constexpr size_t OFF_VALS = OFF_VF   + SZ_VF;
constexpr size_t OFF_H1   = OFF_VALS + SZ_VALS;

// ---------------------------------------------------------------------------
// WMMA fragment loaders (wave32), valid for both global and LDS pointers.
// A 16x32 bf16: lane -> row M=lane&15; half=lane>>4; VGPR v: K=half*16 + 2v,2v+1.
// B 32x16 bf16: lane -> col N=lane&15; VGPR v: K=(lane>>4)*16 + 2v,2v+1.
// Sources are K-contiguous row-major, so fragments are plain b128 fetches.
// ---------------------------------------------------------------------------
DEV v16bf frag_a(const __bf16* p, int ld, int row0, int k0, int lane) {
  const int r = lane & 15, hh = lane >> 4;
  const __bf16* q = p + (size_t)(row0 + r) * ld + (k0 + hh * 16);
  union { v16bf v; uint4 u[2]; } t;
  t.u[0] = *(const uint4*)(q);
  t.u[1] = *(const uint4*)(q + 8);
  return t.v;
}
DEV v16bf frag_b(const __bf16* p, int ld, int col0, int k0, int lane) {
  const int n = lane & 15, hh = lane >> 4;
  const __bf16* q = p + (size_t)(col0 + n) * ld + (k0 + hh * 16);
  union { v16bf v; uint4 u[2]; } t;
  t.u[0] = *(const uint4*)(q);
  t.u[1] = *(const uint4*)(q + 8);
  return t.v;
}
DEV v8f wmma_bf16(v16bf a, v16bf b, v8f c) {
  return __builtin_amdgcn_wmma_f32_16x16x32_bf16(false, a, false, b,
                                                 (short)0, c, false, false);
}

// 16B global->LDS chunk copy (async, ASYNCcnt-tracked)
DEV void copy16(const __bf16* gsrc, __bf16* ldst) {
#if ASYNC_COPY
  __builtin_amdgcn_global_load_async_to_lds_b128((as1_v4i_p)gsrc, (as3_v4i_p)ldst,
                                                 0, 0);
#else
  *(uint4*)ldst = *(const uint4*)gsrc;
#endif
}

// ---------------------------------------------------------------------------
__global__ void cvt_kernel(const float* __restrict__ src, __bf16* __restrict__ dst,
                           int n) {
  for (int i = blockIdx.x * blockDim.x + threadIdx.x; i < n;
       i += gridDim.x * blockDim.x)
    dst[i] = (__bf16)src[i];
}

// ---------------------------------------------------------------------------
// NT GEMM: C[m,n] = sum_k A[m,k]*B[n,k] (+bias, epilogue).
// Block (256 thr / 8 waves) owns a 128x64 macro-tile; waves in a 4x2 grid each
// compute 32x32 (4 WMMA accumulators). K consumed in 32-slabs staged in LDS,
// double-buffered via async global->LDS copies. LDS rows padded to 80B
// (20 dwords = 5 x 16B, 5 coprime with 16) so ds_load_b128 frags are conflict-free.
// EPI 0: QKV scatter -> Q(bf16)/K(bf16)/V(f32) per-head layouts
// EPI 1: bias + ReLU -> bf16     EPI 2: bias -> f32
// ---------------------------------------------------------------------------
template<int KDIM, int NBLK, int EPI>
__global__ __launch_bounds__(256) void gemm_nt_kernel(
    const __bf16* __restrict__ A, const __bf16* __restrict__ Bm,
    const float* __restrict__ bias, void* __restrict__ outp,
    __bf16* __restrict__ Qb, __bf16* __restrict__ Kb, float* __restrict__ Vf) {
  constexpr int BM = 128, BN = 64, BK = 32;
  constexpr int LR = 40;                       // padded LDS row (bf16 elems)
  __shared__ __bf16 shA[2][BM * LR];           // 20 KB
  __shared__ __bf16 shB[2][BN * LR];           // 10 KB

  const int tid  = threadIdx.x;
  const int lane = tid & 31;
  const int wv   = tid >> 5;
  const int wm   = wv >> 1, wn = wv & 1;       // 4x2 wave grid
  const int bm0  = (blockIdx.x / NBLK) * BM;
  const int bn0  = (blockIdx.x % NBLK) * BN;

  // stage one 32-wide K slab into LDS buffer `buf` (3 x 16B chunks per thread)
  auto stage = [&](int k0, int buf) {
#pragma unroll
    for (int it = 0; it < 2; ++it) {           // A: 128 rows x 4 chunks
      const int c = tid + it * 256, row = c >> 2, ch = c & 3;
      copy16(A + (size_t)(bm0 + row) * KDIM + k0 + ch * 8,
             &shA[buf][row * LR + ch * 8]);
    }
    {                                          // B: 64 rows x 4 chunks
      const int row = tid >> 2, ch = tid & 3;
      copy16(Bm + (size_t)(bn0 + row) * KDIM + k0 + ch * 8,
             &shB[buf][row * LR + ch * 8]);
    }
  };

  v8f acc00 = {}, acc01 = {}, acc10 = {}, acc11 = {};
  stage(0, 0);
  int buf = 0;
  for (int k0 = 0; k0 < KDIM; k0 += BK, buf ^= 1) {
    if (k0 + BK < KDIM) {                      // prefetch next slab, then wait
      stage(k0 + BK, buf ^ 1);                 // for current slab only
      WAIT_ASYNC(3);
    } else {
      WAIT_ASYNC(0);
    }
    __syncthreads();                           // slab visible to all waves

    v16bf a0 = frag_a(shA[buf], LR, wm * 32,      0, lane);
    v16bf a1 = frag_a(shA[buf], LR, wm * 32 + 16, 0, lane);
    v16bf b0 = frag_b(shB[buf], LR, wn * 32,      0, lane);
    v16bf b1 = frag_b(shB[buf], LR, wn * 32 + 16, 0, lane);
    acc00 = wmma_bf16(a0, b0, acc00);
    acc01 = wmma_bf16(a0, b1, acc01);
    acc10 = wmma_bf16(a1, b0, acc10);
    acc11 = wmma_bf16(a1, b1, acc11);

    __syncthreads();                           // reads done before overwrite
  }

  const int hh = lane >> 4, cl = lane & 15;
  v8f accs[4] = {acc00, acc01, acc10, acc11};
#pragma unroll
  for (int t = 0; t < 4; ++t) {
    const int mi = (t >> 1) * 16, ni = (t & 1) * 16;
#pragma unroll
    for (int i = 0; i < 8; ++i) {
      const int row = bm0 + wm * 32 + mi + i + 8 * hh;
      const int col = bn0 + wn * 32 + ni + cl;
      float c = accs[t][i] + bias[col];
      if (EPI == 0) {
        const int head = col / (3 * HD), w = col % (3 * HD);
        const int bi = row >> 11, si = row & (Ssz - 1);
        const size_t base = ((size_t)(bi * H + head) * Ssz + si) * HD;
        if (w < HD)          Qb[base + w]            = (__bf16)c;
        else if (w < 2 * HD) Kb[base + (w - HD)]     = (__bf16)c;
        else                 Vf[base + (w - 2 * HD)] = c;
      } else if (EPI == 1) {
        c = fmaxf(c, 0.0f);
        ((__bf16*)outp)[(size_t)row * (NBLK * BN) + col] = (__bf16)c;
      } else {
        ((float*)outp)[(size_t)row * (NBLK * BN) + col] = c;
      }
    }
  }
}

// ---------------------------------------------------------------------------
// Attention: one wave per 16-query tile of one (b,h); 128 key tiles x 2 WMMAs.
// Next K-tile fragment loads issued before the scan (latency hidden by the
// exp/top-k VALU work). Scan split: lanes 0-15 run the sorted top-8 insert
// for row r, lanes 16-31 run the online (max,sumexp) recurrence for row r
// (same LDS addresses -> broadcast reads). (m,l) merged via LDS at the end.
// ---------------------------------------------------------------------------
__global__ __launch_bounds__(128) void attn_topk_kernel(
    const __bf16* __restrict__ Qb, const __bf16* __restrict__ Kb,
    const float* __restrict__ Vf, __bf16* __restrict__ valsb) {
  __shared__ float ltile[4][16 * 16];
  __shared__ float pval[4][16 * TK];
  __shared__ int   pidx[4][16 * TK];
  __shared__ float mstat[4][16];
  __shared__ float lstat[4][16];

  const int lane   = threadIdx.x & 31;
  const int wlocal = threadIdx.x >> 5;
  const int wid    = blockIdx.x * 4 + wlocal;
  const int bh     = wid >> 7;          // wid / (Ssz/16)
  const int qt     = wid & 127;
  const int m0     = bh * Ssz + qt * 16;
  const int hh = lane >> 4, cl = lane & 15;

  const v16bf qa0 = frag_a(Qb, HD, m0, 0,  lane);   // Q tile resident
  const v16bf qa1 = frag_a(Qb, HD, m0, 32, lane);

  float mrun = -3.4e38f, lrun = 0.0f;   // live in lanes 16-31
  float tv[TK]; int ti[TK];             // live in lanes 0-15
#pragma unroll
  for (int j = 0; j < TK; ++j) { tv[j] = -3.4e38f; ti[j] = 0; }

  v16bf kb0 = frag_b(Kb, HD, bh * Ssz, 0,  lane);   // pipeline prologue
  v16bf kb1 = frag_b(Kb, HD, bh * Ssz, 32, lane);

  constexpr int NT = Ssz / 16;
  for (int kt = 0; kt < NT; ++kt) {
    v8f acc = {};
    acc = wmma_bf16(qa0, kb0, acc);
    acc = wmma_bf16(qa1, kb1, acc);
    if (kt + 1 < NT) {                  // issue next tile loads early; their
      const int n1 = bh * Ssz + (kt + 1) * 16;      // wait lands after scan
      kb0 = frag_b(Kb, HD, n1, 0,  lane);
      kb1 = frag_b(Kb, HD, n1, 32, lane);
    }
#pragma unroll
    for (int i = 0; i < 8; ++i)
      ltile[wlocal][(i + 8 * hh) * 16 + cl] = acc[i] * 0.125f;  // /sqrt(HD)
    __syncthreads();

    if (lane < 16) {                    // top-8 sorted-descending insert, row=lane
      const float* rowp = &ltile[wlocal][lane * 16];
#pragma unroll
      for (int j = 0; j < 16; ++j) {
        const float v = rowp[j];
        if (v > tv[TK - 1]) {
          tv[TK - 1] = v; ti[TK - 1] = kt * 16 + j;
#pragma unroll
          for (int p = TK - 1; p > 0; --p) {
            if (tv[p] > tv[p - 1]) {
              float tf = tv[p]; tv[p] = tv[p - 1]; tv[p - 1] = tf;
              int   tt = ti[p]; ti[p] = ti[p - 1]; ti[p - 1] = tt;
            }
          }
        }
      }
    } else {                            // online (max,sumexp), row = lane-16
      const float* rowp = &ltile[wlocal][(lane - 16) * 16];
#pragma unroll
      for (int j = 0; j < 16; ++j) {
        const float v = rowp[j];
        if (v > mrun) { lrun = lrun * __expf(mrun - v) + 1.0f; mrun = v; }
        else          { lrun += __expf(v - mrun); }
      }
    }
    __syncthreads();
  }

  if (lane >= 16) {                     // publish softmax stats
    mstat[wlocal][lane - 16] = mrun;
    lstat[wlocal][lane - 16] = lrun;
  }
  __syncthreads();
  if (lane < 16) {
    const float mrow = mstat[wlocal][lane];
    const float inv  = 1.0f / lstat[wlocal][lane];
#pragma unroll
    for (int j = 0; j < TK; ++j) {
      pval[wlocal][lane * TK + j] = __expf(tv[j] - mrow) * inv;
      pidx[wlocal][lane * TK + j] = ti[j];
    }
  }
  __syncthreads();

  const int bi = bh / H, head = bh % H;
  const size_t vbase = (size_t)bh * Ssz * HD;
  for (int e = lane; e < 16 * TK * HD; e += 32) {
    const int d = e & (HD - 1);
    const int j = (e >> 6) & (TK - 1);
    const int r = e >> 9;
    const float p  = pval[wlocal][r * TK + j];
    const int   ki = pidx[wlocal][r * TK + j];
    const float val = p * Vf[vbase + (size_t)ki * HD + d];
    const size_t orow = (size_t)bi * Ssz + (qt * 16 + r);
    valsb[orow * DV + head * (TK * HD) + j * HD + d] = (__bf16)val;
  }
}

// ---------------------------------------------------------------------------
extern "C" void kernel_launch(void* const* d_in, const int* in_sizes, int n_in,
                              void* d_out, int out_size, void* d_ws, size_t ws_size,
                              hipStream_t stream) {
  const float* x    = (const float*)d_in[0];
  const float* Wqkv = (const float*)d_in[1];
  const float* bqkv = (const float*)d_in[2];
  const float* W1   = (const float*)d_in[3];
  const float* b1   = (const float*)d_in[4];
  const float* W2   = (const float*)d_in[5];
  const float* b2   = (const float*)d_in[6];

  char* ws = (char*)d_ws;
  __bf16* xb    = (__bf16*)(ws + OFF_XB);
  __bf16* wqkvb = (__bf16*)(ws + OFF_WQKV);
  __bf16* w1b   = (__bf16*)(ws + OFF_W1);
  __bf16* w2b   = (__bf16*)(ws + OFF_W2);
  __bf16* Qb    = (__bf16*)(ws + OFF_QB);
  __bf16* Kb    = (__bf16*)(ws + OFF_KB);
  float*  Vf    = (float*) (ws + OFF_VF);
  __bf16* valsb = (__bf16*)(ws + OFF_VALS);
  __bf16* h1b   = (__bf16*)(ws + OFF_H1);

  auto cvt = [&](const float* s, __bf16* d, int n) {
    int blocks = (n + 255) / 256;
    if (blocks > 8192) blocks = 8192;
    cvt_kernel<<<blocks, 256, 0, stream>>>(s, d, n);
  };
  cvt(x,    xb,    M * E);
  cvt(Wqkv, wqkvb, NQKV * E);
  cvt(W1,   w1b,   D1 * DV);
  cvt(W2,   w2b,   E * D1);

  // QKV projection -> per-head Q/K(bf16), V(f32)
  gemm_nt_kernel<E, NQKV / 64, 0>
      <<<(M / 128) * (NQKV / 64), 256, 0, stream>>>(xb, wqkvb, bqkv, nullptr,
                                                    Qb, Kb, Vf);
  // attention logits (WMMA) + online softmax + top-8 + V gather
  attn_topk_kernel<<<(Bsz * H * (Ssz / 16)) / 4, 128, 0, stream>>>(Qb, Kb, Vf,
                                                                   valsb);
  // MLP hidden: relu(vals @ W1^T + b1) -> bf16
  gemm_nt_kernel<DV, D1 / 64, 1>
      <<<(M / 128) * (D1 / 64), 256, 0, stream>>>(valsb, w1b, b1, h1b,
                                                  nullptr, nullptr, nullptr);
  // output: h1 @ W2^T + b2 -> f32
  gemm_nt_kernel<D1, E / 64, 2>
      <<<(M / 128) * (E / 64), 256, 0, stream>>>(h1b, w2b, b2, d_out,
                                                 nullptr, nullptr, nullptr);
}
